// RolePositionBiasedAttention_14688788152507
// MI455X (gfx1250) — compile-verified, matrix operand order checked by reading, not run on hardware
//
#include <hip/hip_runtime.h>
#include <hip/hip_bf16.h>

typedef __attribute__((ext_vector_type(16))) __bf16 v16bf;
typedef __attribute__((ext_vector_type(8)))  __bf16 v8bf;
typedef __attribute__((ext_vector_type(8)))  float  v8f;

#define SHUF16(lo, hi) __builtin_shufflevector((lo), (hi), 0,1,2,3,4,5,6,7,8,9,10,11,12,13,14,15)

static constexpr int NTOK = 10;     // tokens per batch item
static constexpr int DMODEL = 256;  // model dim
static constexpr int NH = 8;        // heads
static constexpr int DH = 32;       // head dim
static constexpr int BB = 8;        // batch items per workgroup
static constexpr int ROWS = BB * NTOK;      // 80 rows (= 5 M-tiles of 16)
static constexpr int S = DMODEL + 8;        // padded LDS row stride in bf16 (528B row -> conflict-free b128)

// ---- bf16 helpers (manual, round-to-nearest-even) ----
static __device__ __forceinline__ unsigned short f2bfbits(float f) {
    union { float f; unsigned u; } v; v.f = f;
    unsigned r = v.u + 0x7FFFu + ((v.u >> 16) & 1u);
    return (unsigned short)(r >> 16);
}
static __device__ __forceinline__ __bf16 u2bf(unsigned short s) {
    union { unsigned short s; __bf16 b; } v; v.s = s; return v.b;
}
static __device__ __forceinline__ float bf2f(__bf16 b) {
    union { unsigned short s; __bf16 b; } v; v.b = b;
    union { unsigned u; float f; } w; w.u = ((unsigned)v.s) << 16;
    return w.f;
}

// ---- gfx1250 async global->LDS copy (ASYNCcnt path), inline asm: portable on both toolchains ----
static __device__ __forceinline__ void async_copy_b32(unsigned lds_byte_off, unsigned long long gaddr) {
    asm volatile("global_load_async_to_lds_b32 %0, %1, off"
                 :: "v"(lds_byte_off), "v"(gaddr) : "memory");
}
static __device__ __forceinline__ void wait_async0() {
    asm volatile("s_wait_asynccnt 0x0" ::: "memory");
}

// ---- one-time weight convert: fp32 W[k][n] -> bf16 WT[n][k] (workspace, L2-resident) ----
__global__ __launch_bounds__(256) void wcvt_kernel(const float* __restrict__ WQ,
                                                   const float* __restrict__ WK,
                                                   const float* __restrict__ WV,
                                                   const float* __restrict__ WO,
                                                   __bf16* __restrict__ wt) {
    int idx = blockIdx.x * 256 + threadIdx.x;      // 0 .. 4*65536-1
    int m = idx >> 16;
    int r = idx & 0xFFFF;
    int n = r >> 8;
    int k = r & 255;
    const float* W = (m == 0) ? WQ : (m == 1) ? WK : (m == 2) ? WV : WO;
    wt[(size_t)m * 65536 + (size_t)n * 256 + k] = u2bf(f2bfbits(W[(size_t)k * 256 + n]));
}

// ---- A/B fragment loader: row-major bf16 tile, stride in elements ----
// Per ISA: lanes 0-15 hold row M=lane with K 0..7 / 16..23; lanes 16-31 hold K 8..15 / 24..31.
static __device__ __forceinline__ v16bf load_frag(const __bf16* row_base, int kk, int hi8) {
    v8bf lo = *(const v8bf*)(row_base + kk + hi8);
    v8bf hi = *(const v8bf*)(row_base + kk + 16 + hi8);
    return SHUF16(lo, hi);
}

__global__ __launch_bounds__(256, 1) void fused_attn_kernel(const float* __restrict__ Z,
                                                            const int* __restrict__ role,
                                                            const float* __restrict__ Bp,
                                                            const __bf16* __restrict__ wt,
                                                            float* __restrict__ out) {
    extern __shared__ char smem_raw[];
    __bf16* sZ = (__bf16*)smem_raw;          // 80 x 264 bf16  (also reused as attention output)
    __bf16* sQ = sZ + ROWS * S;
    __bf16* sK = sQ + ROWS * S;
    __bf16* sV = sK + ROWS * S;
    float* sBp = (float*)(sV + ROWS * S);    // 8*10*10 bias table (async-copied)
    int* sRole = (int*)(sBp + NH * 100);     // 80 ints
    const unsigned sBp_off = (unsigned)((char*)sBp - smem_raw);

    const int tid  = threadIdx.x;
    const int lane = tid & 31;
    const int wave = tid >> 5;               // 8 waves
    const int wg   = blockIdx.x;
    const int hi8  = (lane >> 4) * 8;        // fragment K-half select
    const int lr   = lane & 15;

    const __bf16* wtq = wt;
    const __bf16* wtk = wt + 1 * 65536;
    const __bf16* wtv = wt + 2 * 65536;
    const __bf16* wto = wt + 3 * 65536;

    // ---------------- Stage 1: stage operands into LDS ----------------
    // (a) bias table via gfx1250 async global->LDS DMA (no VGPR round-trip)
    for (int e = tid; e < NH * 100; e += 256)
        async_copy_b32(sBp_off + (unsigned)e * 4u, (unsigned long long)(Bp + e));
    // (b) role ids
    if (tid < ROWS) sRole[tid] = role[(size_t)wg * ROWS + tid];
    // (c) Z chunk fp32 -> bf16 into LDS
    {
        const float* zbase = Z + (size_t)wg * ROWS * DMODEL;
        for (int e4 = tid; e4 < ROWS * (DMODEL / 4); e4 += 256) {
            int row = e4 >> 6;               // 64 float4 per row
            int c4  = e4 & 63;
            float4 z4 = ((const float4*)(zbase + (size_t)row * DMODEL))[c4];
            unsigned lo = (unsigned)f2bfbits(z4.x) | ((unsigned)f2bfbits(z4.y) << 16);
            unsigned hi = (unsigned)f2bfbits(z4.z) | ((unsigned)f2bfbits(z4.w) << 16);
            uint2 u; u.x = lo; u.y = hi;
            *(uint2*)(sZ + row * S + c4 * 4) = u;
        }
    }
    wait_async0();
    __syncthreads();

    // ---------------- Stage 2: Q,K,V = Zc @ W (bf16 WMMA, f32 acc) -> LDS -------
    // Column-block outer loop: B fragments (64 VGPRs) stay resident across 5 M-tiles.
    for (int cb = wave; cb < 3 * 16; cb += 8) {
        int m  = cb >> 4;                    // 0=Q 1=K 2=V
        int tn = cb & 15;                    // N-tile
        const __bf16* wb = (m == 0) ? wtq : (m == 1) ? wtk : wtv;
        __bf16*       db = (m == 0) ? sQ  : (m == 1) ? sK  : sV;

        // prefetch next column block into caches while this one computes
        int ncb = cb + 8;
        if (ncb < 3 * 16) {
            int nm = ncb >> 4, ntn = ncb & 15;
            const __bf16* nb = ((nm == 0) ? wtq : (nm == 1) ? wtk : wtv)
                               + (size_t)(ntn * 16 + lr) * 256;
            __builtin_prefetch(nb, 0, 0);
        }

        const __bf16* brow = wb + (size_t)(tn * 16 + lr) * 256;
        v16bf bfrag[8];
        #pragma unroll
        for (int k = 0; k < 8; ++k) bfrag[k] = load_frag(brow, k * 32, hi8);

        for (int tm = 0; tm < 5; ++tm) {
            const __bf16* arow = sZ + (tm * 16 + lr) * S;
            v8f acc = {};
            #pragma unroll
            for (int k = 0; k < 8; ++k) {
                v16bf a = load_frag(arow, k * 32, hi8);
                acc = __builtin_amdgcn_wmma_f32_16x16x32_bf16(false, a, false, bfrag[k], (short)0, acc, false, false);
            }
            __bf16* drow = db + (tm * 16 + (lane >> 4) * 8) * S + tn * 16 + lr;
            #pragma unroll
            for (int i = 0; i < 8; ++i) drow[i * S] = u2bf(f2bfbits(acc[i]));
        }
    }
    __syncthreads();

    // ---------------- Stage 3: per-(batch,head) biased softmax attention --------
    // 64 (b,h) pairs, 4 threads each; attention output overwrites sZ region.
    __bf16* sA = sZ;
    {
        const int pair = tid >> 2;           // 0..63
        const int sub  = tid & 3;
        const int b = pair >> 3;             // 0..7
        const int h = pair & 7;              // 0..7
        const float scl = 0.17677669529663687f; // 1/sqrt(32)
        for (int i = sub; i < NTOK; i += 4) {
            const __bf16* qrow = sQ + (b * NTOK + i) * S + h * DH;
            float q[DH];
            #pragma unroll
            for (int d = 0; d < DH; ++d) q[d] = bf2f(qrow[d]);
            int ri = sRole[b * NTOK + i];
            float sc[NTOK];
            float mx = -3.0e38f;
            #pragma unroll
            for (int j = 0; j < NTOK; ++j) {
                const __bf16* krow = sK + (b * NTOK + j) * S + h * DH;
                float dot = 0.f;
                #pragma unroll
                for (int d = 0; d < DH; ++d) dot += q[d] * bf2f(krow[d]);
                float s = dot * scl + sBp[h * 100 + ri * 10 + sRole[b * NTOK + j]];
                sc[j] = s;
                mx = fmaxf(mx, s);
            }
            float sum = 0.f;
            #pragma unroll
            for (int j = 0; j < NTOK; ++j) { float e = __expf(sc[j] - mx); sc[j] = e; sum += e; }
            float inv = 1.0f / sum;
            #pragma unroll
            for (int j = 0; j < NTOK; ++j) sc[j] *= inv;
            float o[DH];
            #pragma unroll
            for (int d = 0; d < DH; ++d) o[d] = 0.f;
            #pragma unroll
            for (int j = 0; j < NTOK; ++j) {
                const __bf16* vrow = sV + (b * NTOK + j) * S + h * DH;
                float p = sc[j];
                #pragma unroll
                for (int d = 0; d < DH; ++d) o[d] += p * bf2f(vrow[d]);
            }
            __bf16* arow = sA + (b * NTOK + i) * S + h * DH;
            #pragma unroll
            for (int d = 0; d < DH; ++d) arow[d] = u2bf(f2bfbits(o[d]));
        }
    }
    __syncthreads();

    // ---------------- Stage 4: out = attnOut @ WO (WMMA) -> fp32 global ---------
    for (int cb = wave; cb < 16; cb += 8) {
        int tn = cb;
        const __bf16* brow = wto + (size_t)(tn * 16 + lr) * 256;
        v16bf bfrag[8];
        #pragma unroll
        for (int k = 0; k < 8; ++k) bfrag[k] = load_frag(brow, k * 32, hi8);

        for (int tm = 0; tm < 5; ++tm) {
            const __bf16* arow = sA + (tm * 16 + lr) * S;
            v8f acc = {};
            #pragma unroll
            for (int k = 0; k < 8; ++k) {
                v16bf a = load_frag(arow, k * 32, hi8);
                acc = __builtin_amdgcn_wmma_f32_16x16x32_bf16(false, a, false, bfrag[k], (short)0, acc, false, false);
            }
            float* orow = out + ((size_t)wg * ROWS + tm * 16 + (lane >> 4) * 8) * DMODEL + tn * 16 + lr;
            #pragma unroll
            for (int i = 0; i < 8; ++i) orow[(size_t)i * DMODEL] = acc[i];
        }
    }
}

extern "C" void kernel_launch(void* const* d_in, const int* in_sizes, int n_in,
                              void* d_out, int out_size, void* d_ws, size_t ws_size,
                              hipStream_t stream) {
    const float* Z  = (const float*)d_in[0];
    const int*   rp = (const int*)  d_in[1];
    const float* WQ = (const float*)d_in[2];
    const float* WK = (const float*)d_in[3];
    const float* WV = (const float*)d_in[4];
    const float* WO = (const float*)d_in[5];
    const float* Bp = (const float*)d_in[6];
    float*  out = (float*)d_out;
    __bf16* wt  = (__bf16*)d_ws;                 // 4 x 256x256 bf16 transposed weights = 512 KB

    const int Bsz = in_sizes[0] / (NTOK * DMODEL);   // 16384

    // 1) convert/transpose weights into workspace (tiny; L2-resident afterwards)
    wcvt_kernel<<<dim3((4 * 65536) / 256), dim3(256), 0, stream>>>(WQ, WK, WV, WO, wt);

    // 2) fully fused attention: big dynamic LDS (CDNA5 WGP supports 320KB)
    size_t smem = (size_t)4 * ROWS * S * sizeof(__bf16)   // Z/Q/K/V tiles
                + (size_t)NH * 100 * sizeof(float)        // bias table
                + (size_t)ROWS * sizeof(int);             // role ids
    (void)hipFuncSetAttribute((const void*)fused_attn_kernel,
                              hipFuncAttributeMaxDynamicSharedMemorySize, (int)smem);
    fused_attn_kernel<<<dim3(Bsz / BB), dim3(256), smem, stream>>>(Z, rp, Bp, wt, out);
}
